// HetGNN_86363202388166
// MI455X (gfx1250) — compile-verified, matrix operand order checked by reading, not run on hardware
//
#include <hip/hip_runtime.h>
#include <hip/hip_bf16.h>
#include <math.h>

// ---------------------------------------------------------------------------
// Problem constants (from the reference)
// ---------------------------------------------------------------------------
#define NP   50000
#define NA   20000
#define NE   300000
#define H    256
#define OUTC 16
#define INP  256
#define INA  128

// ---------------------------------------------------------------------------
// WMMA types (CDNA5 / gfx1250, wave32)
// ---------------------------------------------------------------------------
typedef __attribute__((ext_vector_type(16))) __bf16 v16bf;
typedef __attribute__((ext_vector_type(8)))  __bf16 v8bf;
typedef __attribute__((ext_vector_type(4)))  __bf16 v4bf;
typedef __attribute__((ext_vector_type(8)))  float  v8f;

#define BM 64
#define BN 64
#define BK 32
#define LDP (BK + 8)   // padded LDS row stride: 80B (16B-aligned, conflict-free-ish)

__device__ __forceinline__ float gelu_exact(float x) {
    return 0.5f * x * (1.0f + erff(x * 0.70710678118654752440f));
}

// ---------------------------------------------------------------------------
// Tiled GEMM:  C[M,N] = A[M,K] @ B[K,N]  (+ bias, + GELU)
// fp32 in HBM, bf16 through LDS into v_wmma_f32_16x16x32_bf16.
// Requirements (all call sites satisfy): K % 32 == 0, N % 4 == 0,
// A/B/C 16B-aligned.
// FUSE: 0 = none, 1 = +bias, 2 = +bias +GELU
// ---------------------------------------------------------------------------
template <int FUSE>
__global__ __launch_bounds__(256)
void gemm_bf16_wmma(const float* __restrict__ A, const float* __restrict__ B,
                    const float* __restrict__ bias, float* __restrict__ C,
                    int M, int N, int K)
{
    __shared__ __bf16 As[BM][LDP];   // [m][k]
    __shared__ __bf16 Bs[BN][LDP];   // [n][k]  (B staged transposed)

    const int tid  = threadIdx.x;
    const int wave = tid >> 5;         // 0..7
    const int lane = tid & 31;
    const int lo   = lane & 15;
    const int g    = lane >> 4;        // lane half: selects K runs
    const int wm   = wave & 3;         // M-subtile 0..3
    const int wn   = wave >> 2;        // N-half   0..1

    const int m0 = blockIdx.y * BM;
    const int n0 = blockIdx.x * BN;

    const float4* __restrict__ A4 = (const float4*)A;
    const float4* __restrict__ B4 = (const float4*)B;
    const int K4 = K >> 2;
    const int N4 = N >> 2;

    // ---- branch-free staging coordinates (hoisted out of the K loop) ----
    // A: 512 float4-quads per tile; thread handles quads tid and tid+256.
    int a_r0 = (tid) >> 3,        a_q0 = (tid) & 7;
    int a_r1 = (tid + 256) >> 3,  a_q1 = (tid + 256) & 7;
    int a_gm0 = m0 + a_r0, a_gm1 = m0 + a_r1;
    bool a_ok0 = a_gm0 < M, a_ok1 = a_gm1 < M;
    size_t a_row0 = (size_t)(a_ok0 ? a_gm0 : (M - 1)) * K4;
    size_t a_row1 = (size_t)(a_ok1 ? a_gm1 : (M - 1)) * K4;

    // B: quad -> k row kk = qid/16, n-quad nq = qid%16 (coalesced along n).
    int b_k0 = (tid) >> 4,        b_nq0 = (tid) & 15;
    int b_k1 = (tid + 256) >> 4,  b_nq1 = (tid + 256) & 15;
    int b_gn0 = n0 + b_nq0 * 4,   b_gn1 = n0 + b_nq1 * 4;
    bool b_ok0 = b_gn0 < N, b_ok1 = b_gn1 < N;
    int b_col0 = (b_ok0 ? b_gn0 : (N - 4)) >> 2;
    int b_col1 = (b_ok1 ? b_gn1 : (N - 4)) >> 2;

    v8f acc0 = {};
    v8f acc1 = {};

    for (int k0 = 0; k0 < K; k0 += BK) {
        const int k0q = k0 >> 2;

        // ---- issue all 4 b128 global loads up front (no divergence) ----
        float4 va0 = A4[a_row0 + k0q + a_q0];
        float4 va1 = A4[a_row1 + k0q + a_q1];
        float4 vb0 = B4[(size_t)(k0 + b_k0) * N4 + b_col0];
        float4 vb1 = B4[(size_t)(k0 + b_k1) * N4 + b_col1];
        if (!a_ok0) { va0.x = va0.y = va0.z = va0.w = 0.0f; }
        if (!a_ok1) { va1.x = va1.y = va1.z = va1.w = 0.0f; }
        if (!b_ok0) { vb0.x = vb0.y = vb0.z = vb0.w = 0.0f; }
        if (!b_ok1) { vb1.x = vb1.y = vb1.z = vb1.w = 0.0f; }

        // ---- prefetch next K-tile (global_prefetch_b8) ----
        if (k0 + BK < K) {
            __builtin_prefetch(&A4[a_row0 + k0q + 8 + a_q0], 0, 3);
            __builtin_prefetch(&B4[(size_t)(k0 + BK + b_k0) * N4 + b_col0], 0, 3);
        }

        // ---- convert + store to LDS (A: packed b64, B: transposed b16) ----
        v4bf pa0, pa1;
        pa0[0] = (__bf16)va0.x; pa0[1] = (__bf16)va0.y;
        pa0[2] = (__bf16)va0.z; pa0[3] = (__bf16)va0.w;
        pa1[0] = (__bf16)va1.x; pa1[1] = (__bf16)va1.y;
        pa1[2] = (__bf16)va1.z; pa1[3] = (__bf16)va1.w;
        *(v4bf*)&As[a_r0][a_q0 * 4] = pa0;
        *(v4bf*)&As[a_r1][a_q1 * 4] = pa1;

        Bs[b_nq0 * 4 + 0][b_k0] = (__bf16)vb0.x;
        Bs[b_nq0 * 4 + 1][b_k0] = (__bf16)vb0.y;
        Bs[b_nq0 * 4 + 2][b_k0] = (__bf16)vb0.z;
        Bs[b_nq0 * 4 + 3][b_k0] = (__bf16)vb0.w;
        Bs[b_nq1 * 4 + 0][b_k1] = (__bf16)vb1.x;
        Bs[b_nq1 * 4 + 1][b_k1] = (__bf16)vb1.y;
        Bs[b_nq1 * 4 + 2][b_k1] = (__bf16)vb1.z;
        Bs[b_nq1 * 4 + 3][b_k1] = (__bf16)vb1.w;
        __syncthreads();

        // ---- A fragment: lane-half g holds K runs [8g,8g+8) and [16+8g,+8) ----
        int ar = wm * 16 + lo;
        v8bf alo = *(const v8bf*)&As[ar][g * 8];
        v8bf ahi = *(const v8bf*)&As[ar][16 + g * 8];
        v16bf afrag = __builtin_shufflevector(alo, ahi,
                        0,1,2,3,4,5,6,7,8,9,10,11,12,13,14,15);

        // ---- B fragments (two 16-col subtiles), same K-run pattern ----
        int bc0 = wn * 32 + lo;
        v8bf b0lo = *(const v8bf*)&Bs[bc0][g * 8];
        v8bf b0hi = *(const v8bf*)&Bs[bc0][16 + g * 8];
        v16bf bfrag0 = __builtin_shufflevector(b0lo, b0hi,
                        0,1,2,3,4,5,6,7,8,9,10,11,12,13,14,15);

        int bc1 = bc0 + 16;
        v8bf b1lo = *(const v8bf*)&Bs[bc1][g * 8];
        v8bf b1hi = *(const v8bf*)&Bs[bc1][16 + g * 8];
        v16bf bfrag1 = __builtin_shufflevector(b1lo, b1hi,
                        0,1,2,3,4,5,6,7,8,9,10,11,12,13,14,15);

        acc0 = __builtin_amdgcn_wmma_f32_16x16x32_bf16(
                   false, afrag, false, bfrag0, (short)0, acc0, false, false);
        acc1 = __builtin_amdgcn_wmma_f32_16x16x32_bf16(
                   false, afrag, false, bfrag1, (short)0, acc1, false, false);
        __syncthreads();
    }

    // ---- epilogue: C/D layout -> element r maps to row m0+wm*16+8g+r ----
    const int na = n0 + wn * 32 + lo;
    const int nb = na + 16;
    const bool na_ok = na < N;
    const bool nb_ok = nb < N;

    // hoist bias loads out of the r loop (invariant across rows)
    float bias_a = 0.0f, bias_b = 0.0f;
    if (FUSE >= 1) {
        bias_a = bias[na_ok ? na : 0];
        bias_b = bias[nb_ok ? nb : 0];
    }

    #pragma unroll
    for (int r = 0; r < 8; ++r) {
        int m = m0 + wm * 16 + g * 8 + r;
        if (m >= M) continue;
        float va = acc0[r];
        float vb = acc1[r];
        if (na_ok) {
            if (FUSE >= 1) va += bias_a;
            if (FUSE == 2) va = gelu_exact(va);
            C[(size_t)m * N + na] = va;
        }
        if (nb_ok) {
            if (FUSE >= 1) vb += bias_b;
            if (FUSE == 2) vb = gelu_exact(vb);
            C[(size_t)m * N + nb] = vb;
        }
    }
}

// ---------------------------------------------------------------------------
// Workspace zeroing (accumulators + counts, every call: deterministic)
// ---------------------------------------------------------------------------
__global__ void zero_kernel(float* __restrict__ p, size_t n)
{
    size_t i      = (size_t)blockIdx.x * blockDim.x + threadIdx.x;
    size_t stride = (size_t)gridDim.x * blockDim.x;
    for (; i < n; i += stride) p[i] = 0.0f;
}

// ---------------------------------------------------------------------------
// Segment-mean accumulation: per (edge, 4-feature chunk) thread,
// gather msg[src] and atomically add into acc[dst]; one count per edge.
// ---------------------------------------------------------------------------
__global__ __launch_bounds__(256)
void scatter_accum(const float* __restrict__ msg, const int* __restrict__ src,
                   const int* __restrict__ dst, float* __restrict__ acc,
                   float* __restrict__ cnt, int total)
{
    int t = blockIdx.x * 256 + threadIdx.x;
    if (t >= total) return;
    int e = t >> 6;            // H/4 = 64 chunks per edge
    int h = (t & 63) << 2;
    int s = src[e];
    int d = dst[e];
    const float4 v = *(const float4*)(msg + (size_t)s * H + h);
    float* p = acc + (size_t)d * H + h;
    atomicAdd(p + 0, v.x);
    atomicAdd(p + 1, v.y);
    atomicAdd(p + 2, v.z);
    atomicAdd(p + 3, v.w);
    if (h == 0) atomicAdd(cnt + d, 1.0f);
}

// ---------------------------------------------------------------------------
// Per-paper: mean, 2-way attention softmax, combine, residual.
// One 256-thread block per paper row; wave32 shuffle + LDS reduction.
// ---------------------------------------------------------------------------
__global__ __launch_bounds__(256)
void combine_kernel(const float* __restrict__ accw, const float* __restrict__ accc,
                    const float* __restrict__ cntw, const float* __restrict__ cntc,
                    const float* __restrict__ attw, const float* __restrict__ hp,
                    float* __restrict__ out)
{
    int n = blockIdx.x;
    int h = threadIdx.x;               // H == 256 == blockDim.x
    size_t base = (size_t)n * H + h;

    float cw = fmaxf(cntw[n], 1.0f);
    float cc = fmaxf(cntc[n], 1.0f);
    float mw = accw[base] / cw;
    float mc = accc[base] / cc;
    float a  = attw[h];
    float pw = mw * a;
    float pc = mc * a;

    // wave32 reduction
    #pragma unroll
    for (int o = 16; o > 0; o >>= 1) {
        pw += __shfl_down(pw, o, 32);
        pc += __shfl_down(pc, o, 32);
    }
    __shared__ float sw[8], sc[8];
    __shared__ float fw, fc;
    int wv = h >> 5, lane = h & 31;
    if (lane == 0) { sw[wv] = pw; sc[wv] = pc; }
    __syncthreads();
    if (h == 0) {
        float tw = 0.0f, tc = 0.0f;
        #pragma unroll
        for (int i = 0; i < 8; ++i) { tw += sw[i]; tc += sc[i]; }
        float m  = fmaxf(tw, tc);
        float ew = expf(tw - m), ec = expf(tc - m);
        float inv = 1.0f / (ew + ec);
        fw = ew * inv; fc = ec * inv;
    }
    __syncthreads();

    out[base] = fw * mw + fc * mc + hp[base];
}

// ---------------------------------------------------------------------------
// Host-side orchestration
// ---------------------------------------------------------------------------
static inline void launch_gemm(const float* A, const float* B, const float* bias,
                               float* C, int M, int N, int K, int fuse,
                               hipStream_t stream)
{
    dim3 blk(256);
    dim3 grid((N + BN - 1) / BN, (M + BM - 1) / BM);
    if (fuse == 2)      gemm_bf16_wmma<2><<<grid, blk, 0, stream>>>(A, B, bias, C, M, N, K);
    else if (fuse == 1) gemm_bf16_wmma<1><<<grid, blk, 0, stream>>>(A, B, bias, C, M, N, K);
    else                gemm_bf16_wmma<0><<<grid, blk, 0, stream>>>(A, B, nullptr, C, M, N, K);
}

extern "C" void kernel_launch(void* const* d_in, const int* in_sizes, int n_in,
                              void* d_out, int out_size, void* d_ws, size_t ws_size,
                              hipStream_t stream)
{
    const float* paper_x  = (const float*)d_in[0];
    const float* author_x = (const float*)d_in[1];
    const float* ep_w1 = (const float*)d_in[2];  const float* ep_b1 = (const float*)d_in[3];
    const float* ep_w2 = (const float*)d_in[4];  const float* ep_b2 = (const float*)d_in[5];
    const float* ea_w1 = (const float*)d_in[6];  const float* ea_b1 = (const float*)d_in[7];
    const float* ea_w2 = (const float*)d_in[8];  const float* ea_b2 = (const float*)d_in[9];
    const float* rel_w = (const float*)d_in[10];
    const float* rel_c = (const float*)d_in[11];
    const float* att_w = (const float*)d_in[12];
    const float* cl_w1 = (const float*)d_in[13]; const float* cl_b1 = (const float*)d_in[14];
    const float* cl_w2 = (const float*)d_in[15]; const float* cl_b2 = (const float*)d_in[16];
    const int* wsrc = (const int*)d_in[17];
    const int* wdst = (const int*)d_in[18];
    const int* csrc = (const int*)d_in[19];
    const int* cdst = (const int*)d_in[20];
    float* out = (float*)d_out;

    // Workspace layout (floats)
    float* ws = (float*)d_ws;
    size_t o = 0;
    float* tmp   = ws + o; o += (size_t)NP * H;   // hidden / r_writes
    float* h_p   = ws + o; o += (size_t)NP * H;
    float* h_a   = ws + o; o += (size_t)NA * H;
    float* r_c   = ws + o; o += (size_t)NP * H;   // r_cites, then h_new
    float* acc_w = ws + o; o += (size_t)NP * H;   // then z (classifier hidden)
    float* acc_c = ws + o; o += (size_t)NP * H;
    float* cnt_w = ws + o; o += (size_t)NP;
    float* cnt_c = ws + o; o += (size_t)NP;

    // 1-2. Paper encoder
    launch_gemm(paper_x, ep_w1, ep_b1, tmp, NP, H, INP, 2, stream);
    launch_gemm(tmp,     ep_w2, ep_b2, h_p, NP, H, H,   1, stream);
    // 3-4. Author encoder
    launch_gemm(author_x, ea_w1, ea_b1, tmp, NA, H, INA, 2, stream);
    launch_gemm(tmp,      ea_w2, ea_b2, h_a, NA, H, H,   1, stream);
    // 5-6. Per-relation linears (r_writes reuses tmp)
    launch_gemm(h_a, rel_w, nullptr, tmp, NA, H, H, 0, stream);
    launch_gemm(h_p, rel_c, nullptr, r_c, NP, H, H, 0, stream);

    // 7. Zero accumulators + counts (contiguous region)
    size_t zn = 2 * (size_t)NP * H + 2 * (size_t)NP;
    zero_kernel<<<2048, 256, 0, stream>>>(acc_w, zn);

    // 8. Segment-sum scatter (writes: author->paper, cites: paper->paper)
    int total = NE * (H / 4);
    int sgrid = (total + 255) / 256;
    scatter_accum<<<sgrid, 256, 0, stream>>>(tmp, wsrc, wdst, acc_w, cnt_w, total);
    scatter_accum<<<sgrid, 256, 0, stream>>>(r_c, csrc, cdst, acc_c, cnt_c, total);

    // 9. Mean + attention softmax + combine + residual -> h_new (reuse r_c)
    combine_kernel<<<NP, 256, 0, stream>>>(acc_w, acc_c, cnt_w, cnt_c, att_w, h_p, r_c);

    // 10-11. Classifier head (z reuses acc_w)
    launch_gemm(r_c,   cl_w1, cl_b1, acc_w, NP, H,    H, 2, stream);
    launch_gemm(acc_w, cl_w2, cl_b2, out,   NP, OUTC, H, 1, stream);
}
